// P3_64364379898241
// MI455X (gfx1250) — matrix operand order repaired and checked by reading, not compiled, not run on hardware
//
#include <hip/hip_runtime.h>
#include <hip/hip_bf16.h>
#include <math.h>

// ---------------- problem constants ----------------
#define NB     64            // batch (graphs per side)
#define NPER   256           // nodes per graph
#define NN     (NB*NPER)     // 16384 nodes
#define EE     (NB*4096)     // 262144 edges
#define ETOT   (EE+NN)       // edges + self loops
#define FEDGE  16
#define ZDIM   1152

typedef __attribute__((ext_vector_type(16))) _Float16 v16h;
typedef __attribute__((ext_vector_type(8)))  float    v8f;

static inline int cdiv_h(int a, int b) { return (a + b - 1) / b; }

// =====================================================================
// WMMA GEMM:  C[M,Nc] = act( A[M,K] @ W[K,Nc] + bias )
// block = 128 threads (4 waves); block tile = 64 rows x 64 cols.
// Each wave computes a 16x64 strip (4 x 16x16 WMMA accumulators).
// W k-slab (32x64) is staged once per block into LDS as f16,
// transposed [col][k] (+2 pad) -> conflict-free ds_load_b32 fragments.
// A fragments are 4x b128 loads per lane.
// Requires: M % 64 == 0, K % 32 == 0, Nc % 64 == 0 (all call sites comply).
// =====================================================================
#define WSH_STRIDE 34     // 32 k-halves + 2 pad
__global__ void gemm_wmma_k(const float* __restrict__ A, const float* __restrict__ W,
                            const float* __restrict__ bias, float* __restrict__ C,
                            int M, int K, int Nc, int act) {
    __shared__ _Float16 Wsh[64 * WSH_STRIDE];
    int tid  = threadIdx.x;
    int lane = tid & 31;
    int wv   = tid >> 5;                          // 0..3
    int j0 = blockIdx.x * 64;
    int i0 = (blockIdx.y * 4 + wv) * 16;
    int half = lane >> 4;
    int r    = lane & 15;
    v8f acc[4];
#pragma unroll
    for (int t = 0; t < 4; ++t) acc[t] = (v8f){0.f,0.f,0.f,0.f,0.f,0.f,0.f,0.f};

    const float* arow = A + (size_t)(i0 + r) * K;
    for (int kk = 0; kk < K; kk += 32) {
        __syncthreads();                          // protect previous slab reads
        // ---- stage W[kk..kk+32, j0..j0+64] -> LDS f16, transposed ----
#pragma unroll
        for (int i = 0; i < 4; ++i) {
            int idx  = tid + i * 128;             // 0..511
            int row  = idx >> 4;                  // k within slab: 0..31
            int colq = idx & 15;                  // quad of 4 cols
            float4 wq = *(const float4*)(W + (size_t)(kk + row) * Nc + j0 + colq * 4);
            int c0 = colq * 4;
            Wsh[(c0 + 0) * WSH_STRIDE + row] = (_Float16)wq.x;
            Wsh[(c0 + 1) * WSH_STRIDE + row] = (_Float16)wq.y;
            Wsh[(c0 + 2) * WSH_STRIDE + row] = (_Float16)wq.z;
            Wsh[(c0 + 3) * WSH_STRIDE + row] = (_Float16)wq.w;
        }
        __syncthreads();
        // ---- A fragment: 16x32 f16 layout (ISA 7.12.2), 4x b128 loads ----
        float4 a0 = *(const float4*)(arow + kk + 8 * half);
        float4 a1 = *(const float4*)(arow + kk + 8 * half + 4);
        float4 a2 = *(const float4*)(arow + kk + 16 + 8 * half);
        float4 a3 = *(const float4*)(arow + kk + 16 + 8 * half + 4);
        v16h af;
        af[0]  = (_Float16)a0.x; af[1]  = (_Float16)a0.y;
        af[2]  = (_Float16)a0.z; af[3]  = (_Float16)a0.w;
        af[4]  = (_Float16)a1.x; af[5]  = (_Float16)a1.y;
        af[6]  = (_Float16)a1.z; af[7]  = (_Float16)a1.w;
        af[8]  = (_Float16)a2.x; af[9]  = (_Float16)a2.y;
        af[10] = (_Float16)a2.z; af[11] = (_Float16)a2.w;
        af[12] = (_Float16)a3.x; af[13] = (_Float16)a3.y;
        af[14] = (_Float16)a3.z; af[15] = (_Float16)a3.w;
        if (kk + 32 < K) __builtin_prefetch(arow + kk + 32, 0, 1);
        // ---- sweep A across 4 B column tiles (B fragments from LDS) ----
#pragma unroll
        for (int t = 0; t < 4; ++t) {
            int colL = t * 16 + r;
            const _Float16* wcol = &Wsh[colL * WSH_STRIDE];
            v16h bf;
#pragma unroll
            for (int v = 0; v < 8; ++v) {
                int kb = 2 * (v & 3) + 16 * (v >> 2) + 8 * half;       // B 32x16 f16 layout
                bf[2 * v]     = wcol[kb];
                bf[2 * v + 1] = wcol[kb + 1];
            }
            acc[t] = __builtin_amdgcn_wmma_f32_16x16x32_f16(false, af, false, bf,
                                                            (short)0, acc[t], false, false);
        }
    }
#pragma unroll
    for (int t = 0; t < 4; ++t) {
        int col = j0 + t * 16 + r;
        float bv = bias ? bias[col] : 0.f;
#pragma unroll
        for (int v = 0; v < 8; ++v) {
            int row = i0 + v + 8 * half;          // C/D layout (ISA 7.12.2)
            float x = acc[t][v] + bv;
            if (act) x = fmaxf(x, 0.f);
            C[(size_t)row * Nc + col] = x;
        }
    }
}

// =====================================================================
// small dense linear (Nc < 16 or K < 32): one thread per output element
// =====================================================================
__global__ void linear_small_k(const float* __restrict__ A, const float* __restrict__ W,
                               const float* __restrict__ bias, float* __restrict__ C,
                               int M, int K, int Nc, int act) {
    int gid = blockIdx.x * blockDim.x + threadIdx.x;
    if (gid >= M * Nc) return;
    int m = gid / Nc, n = gid % Nc;
    float s = bias ? bias[n] : 0.f;
    for (int k2 = 0; k2 < K; ++k2) s += A[(size_t)m * K + k2] * W[(size_t)k2 * Nc + n];
    if (act) s = fmaxf(s, 0.f);
    C[gid] = s;
}

// =====================================================================
// column means of edge_attr [EE,16] -> meanv[16]
// =====================================================================
__global__ void mean16_k(const float* __restrict__ ea, float* __restrict__ meanv) {
    __shared__ float red[256];
    int t = threadIdx.x;
    int c = t & 15, r0 = t >> 4;
    float s = 0.f;
    for (int r = r0; r < EE; r += 16) s += ea[(size_t)r * FEDGE + c];
    red[t] = s;
    __syncthreads();
    if (t < 16) {
        float tot = 0.f;
        for (int i = 0; i < 16; ++i) tot += red[t + 16 * i];
        meanv[t] = tot / (float)EE;
    }
}

// =====================================================================
// CSR-by-dst build
// =====================================================================
__global__ void zero_int_k(int* p, int n) {
    int i = blockIdx.x * blockDim.x + threadIdx.x;
    if (i < n) p[i] = 0;
}
__global__ void csr_count_k(const int* __restrict__ dst, int* __restrict__ deg) {
    int e = blockIdx.x * blockDim.x + threadIdx.x;
    if (e >= ETOT) return;
    int d = (e < EE) ? dst[e] : (e - EE);
    atomicAdd(&deg[d], 1);
}
__global__ void scan_excl_k(const int* __restrict__ deg, int* __restrict__ rowptr) {
    __shared__ int part[256];
    int t = threadIdx.x;
    const int CH = NN / 256;
    int base = t * CH;
    int s = 0;
    for (int i = 0; i < CH; ++i) s += deg[base + i];
    part[t] = s;
    __syncthreads();
    if (t == 0) {
        int acc = 0;
        for (int i = 0; i < 256; ++i) { int v = part[i]; part[i] = acc; acc += v; }
        rowptr[NN] = acc;
    }
    __syncthreads();
    int acc = part[t];
    for (int i = 0; i < CH; ++i) { rowptr[base + i] = acc; acc += deg[base + i]; }
}
__global__ void csr_fill_k(const int* __restrict__ dst, const int* __restrict__ rowptr,
                           int* __restrict__ cursor, int* __restrict__ eidx) {
    int e = blockIdx.x * blockDim.x + threadIdx.x;
    if (e >= ETOT) return;
    int d = (e < EE) ? dst[e] : (e - EE);
    int pos = atomicAdd(&cursor[d], 1);
    eidx[rowptr[d] + pos] = e;
}

// =====================================================================
// GATv2 edge logits: logit[e,h] = sum_c att[h,c]*lrelu(xl[s,h,c]+xr[d,h,c]+ee[h,c])
// one thread per (edge, head); ee computed on the fly (We is L2-hot)
// =====================================================================
__global__ void edge_logits_k(const float* __restrict__ xl, const float* __restrict__ xr,
                              const float* __restrict__ ea, const float* __restrict__ meanv,
                              const int* __restrict__ src, const int* __restrict__ dst,
                              const float* __restrict__ We, const float* __restrict__ att,
                              float* __restrict__ logit, int H, int OC) {
    long gid = (long)blockIdx.x * blockDim.x + threadIdx.x;
    if (gid >= (long)ETOT * H) return;
    int e = (int)(gid / H);
    int h = (int)(gid % H);
    int s, d;
    const float* er;
    if (e < EE) { s = src[e]; d = dst[e]; er = ea + (size_t)e * FEDGE; }
    else        { s = e - EE; d = s;      er = meanv; }
    float erow[FEDGE];
#pragma unroll
    for (int t = 0; t < FEDGE; ++t) erow[t] = er[t];
    int hc = H * OC;
    const float* xls = xl + (size_t)s * hc + h * OC;
    const float* xrd = xr + (size_t)d * hc + h * OC;
    const float* wc  = We + h * OC;
    const float* ah  = att + h * OC;
    float lg = 0.f;
    for (int c = 0; c < OC; ++c) {
        float eec = 0.f;
#pragma unroll
        for (int t = 0; t < FEDGE; ++t) eec += erow[t] * wc[(size_t)t * hc + c];
        float v = xls[c] + xrd[c] + eec;
        v = (v > 0.f) ? v : 0.2f * v;
        lg += v * ah[c];
    }
    logit[(size_t)e * H + h] = lg;
}

// =====================================================================
// GATv2 per-dst softmax + aggregation (CSR, no atomics)
// grid = N*H blocks, block = OC threads (channel)
// =====================================================================
__global__ void gat_agg_k(const float* __restrict__ xl, const float* __restrict__ logit,
                          const int* __restrict__ rowptr, const int* __restrict__ eidx,
                          const int* __restrict__ src, const float* __restrict__ bias,
                          float* __restrict__ out, int H, int OC, int act) {
    int d = blockIdx.x / H;
    int h = blockIdx.x % H;
    int c = threadIdx.x;
    int beg = rowptr[d], end = rowptr[d + 1];
    float mx = -1e30f;
    for (int i = beg; i < end; ++i) mx = fmaxf(mx, logit[(size_t)eidx[i] * H + h]);
    float den = 0.f;
    for (int i = beg; i < end; ++i) den += expf(logit[(size_t)eidx[i] * H + h] - mx);
    int hc = H * OC;
    float acc = 0.f;
    for (int i = beg; i < end; ++i) {
        int e = eidx[i];
        float a = expf(logit[(size_t)e * H + h] - mx) / den;
        int s = (e < EE) ? src[e] : (e - EE);
        acc += a * xl[(size_t)s * hc + h * OC + c];
    }
    float o = acc + bias[h * OC + c];
    if (act) o = fmaxf(o, 0.f);
    out[(size_t)d * hc + h * OC + c] = o;
}

// =====================================================================
// layernorm over last dim; block = D threads (D power of 2), grid = rows
// out = (a [+ b] - mean)/sqrt(var+eps) * g + beta
// =====================================================================
__global__ void ln_k(const float* __restrict__ a, const float* __restrict__ b,
                     const float* __restrict__ g, const float* __restrict__ beta,
                     float* __restrict__ out) {
    extern __shared__ float red[];
    int D = blockDim.x, t = threadIdx.x;
    size_t off = (size_t)blockIdx.x * D;
    float x = a[off + t] + (b ? b[off + t] : 0.f);
    red[t] = x; __syncthreads();
    for (int o = D >> 1; o > 0; o >>= 1) { if (t < o) red[t] += red[t + o]; __syncthreads(); }
    float mean = red[0] / D; __syncthreads();
    float dv = x - mean;
    red[t] = dv * dv; __syncthreads();
    for (int o = D >> 1; o > 0; o >>= 1) { if (t < o) red[t] += red[t + o]; __syncthreads(); }
    float var = red[0] / D;
    out[off + t] = dv * rsqrtf(var + 1e-5f) * g[t] + beta[t];
}

// =====================================================================
// cross attention core: one block per (graph,head); K/V tiles in LDS (f16)
// q,k,v,o are [N,256] with channel = h*32 + d
// =====================================================================
__global__ void cross_attn_k(const float* __restrict__ q, const float* __restrict__ k,
                             const float* __restrict__ v, float* __restrict__ o) {
    __shared__ _Float16 Ksh[NPER * 32];
    __shared__ _Float16 Vsh[NPER * 32];
    __shared__ float qsh[32];
    __shared__ float red[NPER];
    __shared__ float ash[NPER];
    int b = blockIdx.x >> 3;
    int h = blockIdx.x & 7;
    int t = threadIdx.x;
    size_t base = (size_t)b * NPER;
    for (int idx = t; idx < NPER * 32; idx += NPER) {
        int row = idx >> 5, dch = idx & 31;
        Ksh[idx] = (_Float16)k[(base + row) * 256 + h * 32 + dch];
        Vsh[idx] = (_Float16)v[(base + row) * 256 + h * 32 + dch];
    }
    __syncthreads();
    for (int qi = 0; qi < NPER; ++qi) {
        if (t < 32) qsh[t] = q[(base + qi) * 256 + h * 32 + t];
        __syncthreads();
        float s = 0.f;
#pragma unroll
        for (int dch = 0; dch < 32; ++dch) s += qsh[dch] * (float)Ksh[t * 32 + dch];
        s *= 0.17677669529663687f;             // 1/sqrt(32)
        red[t] = s; __syncthreads();
        for (int off2 = 128; off2 > 0; off2 >>= 1) {
            if (t < off2) red[t] = fmaxf(red[t], red[t + off2]);
            __syncthreads();
        }
        float mx = red[0]; __syncthreads();
        float ex = expf(s - mx);
        red[t] = ex; __syncthreads();
        for (int off2 = 128; off2 > 0; off2 >>= 1) {
            if (t < off2) red[t] += red[t + off2];
            __syncthreads();
        }
        float den = red[0];
        ash[t] = ex / den; __syncthreads();
        if (t < 32) {
            float accv = 0.f;
            for (int j = 0; j < NPER; ++j) accv += ash[j] * (float)Vsh[j * 32 + t];
            o[(base + qi) * 256 + h * 32 + t] = accv;
        }
        __syncthreads();
    }
}

// =====================================================================
// attention pooling: block per graph (256 threads = nodes = features)
// =====================================================================
__global__ void attn_pool_k(const float* __restrict__ x, const float* __restrict__ gate,
                            float* __restrict__ out) {
    __shared__ float red[NPER];
    __shared__ float ash[NPER];
    int b = blockIdx.x, t = threadIdx.x;
    size_t base = (size_t)b * NPER;
    float g = gate[base + t];
    red[t] = g; __syncthreads();
    for (int o = 128; o > 0; o >>= 1) { if (t < o) red[t] = fmaxf(red[t], red[t + o]); __syncthreads(); }
    float mx = red[0]; __syncthreads();
    float ex = expf(g - mx);
    red[t] = ex; __syncthreads();
    for (int o = 128; o > 0; o >>= 1) { if (t < o) red[t] += red[t + o]; __syncthreads(); }
    float den = red[0];
    ash[t] = ex / den; __syncthreads();
    float acc = 0.f;
    for (int j = 0; j < NPER; ++j) acc += ash[j] * x[(base + j) * 256 + t];
    out[(size_t)b * 256 + t] = acc;
}

// =====================================================================
// BatchNorm-eval (running stats 0/1) + relu, in place
// =====================================================================
__global__ void bn_relu_k(float* __restrict__ x, const float* __restrict__ g,
                          const float* __restrict__ b, int total, int D) {
    int i = blockIdx.x * blockDim.x + threadIdx.x;
    if (i >= total) return;
    int c = i % D;
    float inv = rsqrtf(1.0f + 1e-5f);
    x[i] = fmaxf(x[i] * g[c] * inv + b[c], 0.f);
}

// =====================================================================
// concat z = [dg | ag | dag | aag | dhp | ahp]  -> [B,1152]
// =====================================================================
__global__ void concat_k(const float* dg, const float* ag, const float* dag,
                         const float* aag, const float* dhp, const float* ahp,
                         float* __restrict__ z) {
    int i = blockIdx.x * blockDim.x + threadIdx.x;
    if (i >= NB * ZDIM) return;
    int b = i / ZDIM, c = i % ZDIM;
    float v;
    if      (c < 256)  v = dg [b * 256 + c];
    else if (c < 512)  v = ag [b * 256 + (c - 256)];
    else if (c < 768)  v = dag[b * 256 + (c - 512)];
    else if (c < 1024) v = aag[b * 256 + (c - 768)];
    else if (c < 1088) v = dhp[b * 64  + (c - 1024)];
    else               v = ahp[b * 64  + (c - 1088)];
    z[i] = v;
}

// ---------------- host-side param views ----------------
struct Lin   { const float *w, *b; };
struct Lnp   { const float *g, *b; };
struct Gat   { const float *Wl, *Wr, *We, *att, *bias; };
struct Enc   { Gat c1, c2, c3; Lin g1, g2, r1, r2; };
struct CrossP{ Lin q, k, v, o; Lnp ln1, ln2; Lin fc1, fc2; };

extern "C" void kernel_launch(void* const* d_in, const int* in_sizes, int n_in,
                              void* d_out, int out_size, void* d_ws, size_t ws_size,
                              hipStream_t stream) {
    (void)in_sizes; (void)n_in; (void)out_size; (void)ws_size;

    // ---------- inputs (setup_inputs dict order) ----------
    const float* dx   = (const float*)d_in[0];
    const int*   dsrc = (const int*)  d_in[1];
    const int*   ddst = dsrc + EE;
    const float* dea  = (const float*)d_in[2];
    const float* ax   = (const float*)d_in[3];
    const int*   asrc = (const int*)  d_in[4];
    const int*   adst = asrc + EE;
    const float* aea  = (const float*)d_in[5];
    // d_in[6]/d_in[7] = batch vectors (uniform arange, unused)

    // ---------- params (make_params insertion order) ----------
    int pi = 8;
    auto F       = [&]() { return (const float*)d_in[pi++]; };
    auto getLin  = [&]() { Lin l; l.w = F(); l.b = F(); return l; };
    auto getLn   = [&]() { Lnp l; l.g = F(); l.b = F(); return l; };
    auto getGat  = [&]() { Gat g; g.Wl = F(); g.Wr = F(); g.We = F(); g.att = F(); g.bias = F(); return g; };
    auto getEnc  = [&]() { Enc e; e.c1 = getGat(); e.c2 = getGat(); e.c3 = getGat();
                           e.g1 = getLin(); e.g2 = getLin(); e.r1 = getLin(); e.r2 = getLin(); return e; };
    auto getCross= [&]() { CrossP c; c.q = getLin(); c.k = getLin(); c.v = getLin(); c.o = getLin();
                           c.ln1 = getLn(); c.ln2 = getLn(); c.fc1 = getLin(); c.fc2 = getLin(); return c; };
    Enc    Ed = getEnc(), Ea2 = getEnc();
    CrossP C1 = getCross(), C2 = getCross();
    Lin pool1 = getLin(), pool2 = getLin();
    Lnp ng = getLn(), na = getLn();
    Lin hp = getLin(); Lnp hpln = getLn();
    Lin h1 = getLin(); Lnp bn1 = getLn();
    Lin h2 = getLin(); Lnp bn2 = getLn();
    Lin h3 = getLin();

    // ---------- workspace carve ----------
    char* wp = (char*)d_ws;
    auto carve = [&](size_t bytes) { void* p = (void*)wp; wp += ((bytes + 255) / 256) * 256; return p; };
    float* S0    = (float*)carve((size_t)NN * 512 * 4);
    float* S1    = (float*)carve((size_t)NN * 512 * 4);
    float* S2    = (float*)carve((size_t)NN * 512 * 4);
    float* S3    = (float*)carve((size_t)NN * 512 * 4);
    float* LOGIT = (float*)carve((size_t)ETOT * 8 * 4);
    float* GT    = (float*)carve((size_t)NN * 128 * 4);
    float* GV    = (float*)carve((size_t)NN * 4);
    float* meanv = (float*)carve(16 * 4);
    int*   deg   = (int*)  carve((size_t)NN * 4);         // reused as cursor
    int*   rowp  = (int*)  carve((size_t)(NN + 1) * 4);
    int*   eidx  = (int*)  carve((size_t)ETOT * 4);
    float* dn    = (float*)carve((size_t)NN * 256 * 4);
    float* an    = (float*)carve((size_t)NN * 256 * 4);
    float* d2b   = (float*)carve((size_t)NN * 256 * 4);
    float* a2b   = (float*)carve((size_t)NN * 256 * 4);
    float* dgr   = (float*)carve((size_t)NB * 256 * 4);
    float* agr   = (float*)carve((size_t)NB * 256 * 4);
    float* dagr  = (float*)carve((size_t)NB * 256 * 4);
    float* aagr  = (float*)carve((size_t)NB * 256 * 4);
    float* dgl   = (float*)carve((size_t)NB * 256 * 4);
    float* agl   = (float*)carve((size_t)NB * 256 * 4);
    float* dagl  = (float*)carve((size_t)NB * 256 * 4);
    float* aagl  = (float*)carve((size_t)NB * 256 * 4);
    float* dh2   = (float*)carve((size_t)NB * 2 * 4);
    float* ah2   = (float*)carve((size_t)NB * 2 * 4);
    float* th    = (float*)carve((size_t)NB * 128 * 4);
    float* tmp64 = (float*)carve((size_t)NB * 64 * 4);
    float* dhpb  = (float*)carve((size_t)NB * 64 * 4);
    float* ahpb  = (float*)carve((size_t)NB * 64 * 4);
    float* z     = (float*)carve((size_t)NB * ZDIM * 4);
    float* t1    = (float*)carve((size_t)NB * 512 * 4);
    float* t2    = (float*)carve((size_t)NB * 256 * 4);

    // ---------- launch helpers ----------
    auto gemm = [&](const float* A, const float* W, const float* bias, float* C,
                    int M, int K, int Nc, int act) {
        dim3 g(Nc / 64, cdiv_h(M, 64));
        gemm_wmma_k<<<g, 128, 0, stream>>>(A, W, bias, C, M, K, Nc, act);
    };
    auto ln = [&](const float* a, const float* b, const Lnp& p, float* out, int rows, int D) {
        ln_k<<<rows, D, D * sizeof(float), stream>>>(a, b, p.g, p.b, out);
    };
    auto pool = [&](const float* x, const Lin& g1, const Lin& g2, float* out) {
        gemm(x, g1.w, g1.b, GT, NN, 256, 128, 1);
        linear_small_k<<<cdiv_h(NN, 256), 256, 0, stream>>>(GT, g2.w, g2.b, GV, NN, 128, 1, 0);
        attn_pool_k<<<NB, NPER, 0, stream>>>(x, GV, out);
    };

    // ---------- one GATv2 encoder ----------
    auto run_side = [&](const float* x0, const int* src, const int* dst,
                        const float* ea, const Enc& E, float* emb) {
        mean16_k<<<1, 256, 0, stream>>>(ea, meanv);
        // CSR by dst (shared by all 3 convs)
        zero_int_k<<<cdiv_h(NN, 256), 256, 0, stream>>>(deg, NN);
        csr_count_k<<<cdiv_h(ETOT, 256), 256, 0, stream>>>(dst, deg);
        scan_excl_k<<<1, 256, 0, stream>>>(deg, rowp);
        zero_int_k<<<cdiv_h(NN, 256), 256, 0, stream>>>(deg, NN);   // cursor
        csr_fill_k<<<cdiv_h(ETOT, 256), 256, 0, stream>>>(dst, rowp, deg, eidx);
        // conv1: 64 -> 8x64
        gemm(x0, E.c1.Wl, nullptr, S0, NN, 64, 512, 0);
        gemm(x0, E.c1.Wr, nullptr, S1, NN, 64, 512, 0);
        edge_logits_k<<<cdiv_h(ETOT * 8, 256), 256, 0, stream>>>(S0, S1, ea, meanv, src, dst,
                                                                 E.c1.We, E.c1.att, LOGIT, 8, 64);
        gat_agg_k<<<NN * 8, 64, 0, stream>>>(S0, LOGIT, rowp, eidx, src, E.c1.bias, S2, 8, 64, 1);
        // conv2: 512 -> 8x64
        gemm(S2, E.c2.Wl, nullptr, S0, NN, 512, 512, 0);
        gemm(S2, E.c2.Wr, nullptr, S1, NN, 512, 512, 0);
        edge_logits_k<<<cdiv_h(ETOT * 8, 256), 256, 0, stream>>>(S0, S1, ea, meanv, src, dst,
                                                                 E.c2.We, E.c2.att, LOGIT, 8, 64);
        gat_agg_k<<<NN * 8, 64, 0, stream>>>(S0, LOGIT, rowp, eidx, src, E.c2.bias, S3, 8, 64, 1);
        // conv3: 512 -> 1x256
        gemm(S3, E.c3.Wl, nullptr, S0, NN, 512, 256, 0);
        gemm(S3, E.c3.Wr, nullptr, S1, NN, 512, 256, 0);
        edge_logits_k<<<cdiv_h(ETOT, 256), 256, 0, stream>>>(S0, S1, ea, meanv, src, dst,
                                                             E.c3.We, E.c3.att, LOGIT, 1, 256);
        gat_agg_k<<<NN, 256, 0, stream>>>(S0, LOGIT, rowp, eidx, src, E.c3.bias, emb, 1, 256, 0);
    };

    // ---------- cross attention block ----------
    auto run_cross = [&](const float* sx, const float* tx, const CrossP& C, float* out) {
        gemm(sx, C.q.w, C.q.b, S0, NN, 256, 256, 0);
        gemm(tx, C.k.w, C.k.b, S1, NN, 256, 256, 0);
        gemm(tx, C.v.w, C.v.b, S2, NN, 256, 256, 0);
        cross_attn_k<<<NB * 8, NPER, 0, stream>>>(S0, S1, S2, S3);
        gemm(S3, C.o.w, C.o.b, S0, NN, 256, 256, 0);
        ln(sx, S0, C.ln1, S1, NN, 256);                 // sx1 = lnorm(src + o)
        gemm(S1, C.fc1.w, C.fc1.b, S2, NN, 256, 512, 1);
        gemm(S2, C.fc2.w, C.fc2.b, S0, NN, 512, 256, 0);
        ln(S1, S0, C.ln2, out, NN, 256);                // out = lnorm(sx1 + f)
    };

    // ================= pipeline =================
    run_side(dx, dsrc, ddst, dea, Ed,  dn);
    run_side(ax, asrc, adst, aea, Ea2, an);

    pool(dn, Ed.g1,  Ed.g2,  dgr);
    pool(an, Ea2.g1, Ea2.g2, agr);

    // reg heads on raw pooled vectors
    gemm(dgr, Ed.r1.w, Ed.r1.b, th, NB, 256, 128, 1);
    linear_small_k<<<1, 128, 0, stream>>>(th, Ed.r2.w, Ed.r2.b, dh2, NB, 128, 2, 0);
    gemm(agr, Ea2.r1.w, Ea2.r1.b, th, NB, 256, 128, 1);
    linear_small_k<<<1, 128, 0, stream>>>(th, Ea2.r2.w, Ea2.r2.b, ah2, NB, 128, 2, 0);

    run_cross(dn, an, C1, d2b);
    run_cross(an, dn, C2, a2b);

    pool(d2b, pool1, pool2, dagr);
    pool(a2b, pool1, pool2, aagr);

    ln(dgr,  nullptr, ng, dgl,  NB, 256);
    ln(agr,  nullptr, ng, agl,  NB, 256);
    ln(dagr, nullptr, na, dagl, NB, 256);
    ln(aagr, nullptr, na, aagl, NB, 256);

    linear_small_k<<<cdiv_h(NB * 64, 256), 256, 0, stream>>>(dh2, hp.w, hp.b, tmp64, NB, 2, 64, 1);
    ln(tmp64, nullptr, hpln, dhpb, NB, 64);
    linear_small_k<<<cdiv_h(NB * 64, 256), 256, 0, stream>>>(ah2, hp.w, hp.b, tmp64, NB, 2, 64, 1);
    ln(tmp64, nullptr, hpln, ahpb, NB, 64);

    concat_k<<<cdiv_h(NB * ZDIM, 256), 256, 0, stream>>>(dgl, agl, dagl, aagl, dhpb, ahpb, z);

    gemm(z,  h1.w, h1.b, t1, NB, ZDIM, 512, 0);
    bn_relu_k<<<cdiv_h(NB * 512, 256), 256, 0, stream>>>(t1, bn1.g, bn1.b, NB * 512, 512);
    gemm(t1, h2.w, h2.b, t2, NB, 512, 256, 0);
    bn_relu_k<<<cdiv_h(NB * 256, 256), 256, 0, stream>>>(t2, bn2.g, bn2.b, NB * 256, 256);
    linear_small_k<<<1, 64, 0, stream>>>(t2, h3.w, h3.b, (float*)d_out, NB, 256, 1, 0);
}
